// PostProcessingModule_24601572671998
// MI455X (gfx1250) — matrix operand order repaired and checked by reading, not compiled, not run on hardware
//
#include <hip/hip_runtime.h>
#include <hip/hip_bf16.h>

#ifndef __has_builtin
#define __has_builtin(x) 0
#endif

// ---------------------------------------------------------------------------
// Problem constants (from reference)
// ---------------------------------------------------------------------------
#define BB 16
#define ND 4
#define HH 320
#define WW 480
#define PP (HH * WW)          // 153600 pixels per batch
#define KK 32                 // clusters == wave32 width
#define CONF 0.9f
#define F16_BITS_16 0x41800000u   // bits of 16.0f == MARGIN^2
#define MIN_CLUSTER 15
#define MAX_X 103.0f
#define MPP0 0.2f
#define MPP1 0.2f

typedef __attribute__((ext_vector_type(16))) _Float16 v16h;
typedef __attribute__((ext_vector_type(8)))  float    v8f;
typedef __attribute__((ext_vector_type(8)))  int      int8v;

#if defined(__HIP_DEVICE_COMPILE__) && __has_builtin(__builtin_amdgcn_global_load_async_to_lds_b32)
#define HAVE_ASYNC_LDS 1
typedef __attribute__((address_space(1))) int as1_int;   // "__device__ int"
typedef __attribute__((address_space(3))) int as3_int;   // "__shared__ int"
#else
#define HAVE_ASYNC_LDS 0
#endif

// ---------------------------------------------------------------------------
// Kernel 1: per-batch ordered compaction of valid pixels (seg >= CONF).
// cids[b][p] = -1 init; cidx = original index per valid pixel (row-major
// order preserved); cemb = gathered float4 embedding; nvalid[b].
// ---------------------------------------------------------------------------
__global__ void compact_kernel(const float* __restrict__ seg,
                               const float* __restrict__ emb,
                               int* __restrict__ cids,
                               int* __restrict__ cidx,
                               float4* __restrict__ cemb,
                               int* __restrict__ nvalid) {
  const int b    = blockIdx.x;
  const int tid  = threadIdx.x;
  const int lane = tid & 31;
  const int wave = tid >> 5;

  __shared__ int wsum[8];
  __shared__ int wbase[8];
  __shared__ int run_base;
  if (tid == 0) run_base = 0;
  __syncthreads();

  const float* segb = seg + (size_t)b * PP;
  int*    cids_b = cids + (size_t)b * PP;
  int*    cidx_b = cidx + (size_t)b * PP;
  float4* cemb_b = cemb + (size_t)b * PP;

  for (int chunk = 0; chunk < PP; chunk += 256) {
    const int p = chunk + tid;
    const float s = segb[p];
    const int v = (s >= CONF) ? 1 : 0;
    cids_b[p] = -1;

    int incl = v;                       // inclusive wave32 prefix sum
#pragma unroll
    for (int d = 1; d < 32; d <<= 1) {
      int o = __shfl_up(incl, d, 32);
      if (lane >= d) incl += o;
    }
    if (lane == 31) wsum[wave] = incl;
    __syncthreads();
    if (tid == 0) {
      int acc = run_base;
      for (int w = 0; w < 8; ++w) { wbase[w] = acc; acc += wsum[w]; }
      run_base = acc;
    }
    __syncthreads();
    if (v) {
      const int rank = wbase[wave] + incl - 1;
      cidx_b[rank] = p;
      float4 e;
      e.x = emb[((size_t)b * ND + 0) * PP + p];
      e.y = emb[((size_t)b * ND + 1) * PP + p];
      e.z = emb[((size_t)b * ND + 2) * PP + p];
      e.w = emb[((size_t)b * ND + 3) * PP + p];
      cemb_b[rank] = e;
    }
    __syncthreads();
  }
  if (tid == 0) nvalid[b] = run_base;
}

// ---------------------------------------------------------------------------
// Kernel 2: sequential greedy clustering over compacted valid pixels.
// One wave32 per batch; lane k owns cluster k. Argmin is a single u32
// min-reduce of (dist_bits & ~31)|lane (valid because dist >= 0).
// ---------------------------------------------------------------------------
__global__ void cluster_kernel(const int* __restrict__ nvalid,
                               const int* __restrict__ cidx,
                               const float4* __restrict__ cemb,
                               int* __restrict__ cids,
                               int* __restrict__ counts) {
  const int b    = blockIdx.x;
  const int lane = threadIdx.x;
  const int n = nvalid[b];
  const int*    cidx_b = cidx + (size_t)b * PP;
  const float4* cemb_b = cemb + (size_t)b * PP;
  int* cids_b = cids + (size_t)b * PP;

  float c0 = 0.f, c1 = 0.f, c2 = 0.f, c3 = 0.f;
  int cnt = 0;

  for (int t = 0; t < n; ++t) {
    if ((t & 15) == 0) {
      __builtin_prefetch((const void*)(cemb_b + t + 64), 0, 0);
      __builtin_prefetch((const void*)(cidx_b + t + 256), 0, 0);
    }
    const float4 e = cemb_b[t];   // uniform address: broadcast load

    const float dx = c0 - e.x, dy = c1 - e.y, dz = c2 - e.z, dw = c3 - e.w;
    const float dist = dx * dx + dy * dy + dz * dz + dw * dw;

    unsigned key = (cnt > 0)
        ? ((__float_as_uint(dist) & 0xFFFFFFE0u) | (unsigned)lane)
        : (0x7F800000u | (unsigned)lane);
    unsigned r = key;
#pragma unroll
    for (int off = 16; off >= 1; off >>= 1) {
      unsigned o = (unsigned)__shfl_xor((int)r, off, 32);
      r = (o < r) ? o : r;
    }
    const int bi = (int)(r & 31u);

#if __has_builtin(__builtin_amdgcn_ballot_w32)
    const unsigned am = __builtin_amdgcn_ballot_w32(cnt > 0);
    const int act = __popc(am);
#else
    int act = (cnt > 0) ? 1 : 0;
#pragma unroll
    for (int off = 16; off >= 1; off >>= 1) act += __shfl_xor(act, off, 32);
#endif

    const bool assign = (r < F16_BITS_16);      // min_dist^2 < 16
    const bool go = assign || (act < KK);
    const int  upd = assign ? bi : act;         // create slot = n_active
    if (go && lane == upd) {
      if (assign) {
        const float fc = (float)cnt;
        const float inv = 1.0f / (fc + 1.0f);
        c0 = (c0 * fc + e.x) * inv;
        c1 = (c1 * fc + e.y) * inv;
        c2 = (c2 * fc + e.z) * inv;
        c3 = (c3 * fc + e.w) * inv;
      } else {
        c0 = e.x; c1 = e.y; c2 = e.z; c3 = e.w;
      }
      cnt += 1;
    }
    if (lane == 0) cids_b[cidx_b[t]] = go ? upd : -1;
  }
  counts[b * KK + lane] = cnt;
}

// ---------------------------------------------------------------------------
// Kernel 3a: per-(b,h) segment sums as a one-hot GEMM on WMMA.
//   D[32 x 3] = A[32 x 480] * B[480 x 3]
//   A[k][p] = (cid[p] == k), B[p] = (1, col_p + sigmoid(off_p), z_p)
// Branch-free: B operand = one 32-byte LDS vector load from a per-lane base pointer
// (ones / x / z / zeros table chosen by column). A operand = two int8 LDS
// vector loads + packed cndmask_b16 selects. cids row is staged into LDS
// via GLOBAL_LOAD_ASYNC_TO_LDS (ASYNCcnt) overlapped with sigmoid math.
// ---------------------------------------------------------------------------
__global__ void segsum_wmma_kernel(const int* __restrict__ cids,
                                   const float* __restrict__ off_logits,
                                   const float* __restrict__ height,
                                   float* __restrict__ cnt_kh,
                                   float* __restrict__ sx_kh,
                                   float* __restrict__ sz_kh) {
  const int bh = blockIdx.x;
  const int b = bh / HH;
  const int h = bh % HH;
  const int lane = threadIdx.x;

  __shared__ __align__(16) int      s_cid[WW];
  __shared__ __align__(16) _Float16 s_x[WW];
  __shared__ __align__(16) _Float16 s_z[WW];
  __shared__ __align__(16) _Float16 s_one[WW];
  __shared__ __align__(16) _Float16 s_zero[WW];

  const size_t rowoff = (size_t)b * PP + (size_t)h * WW;

#if HAVE_ASYNC_LDS
  for (int j = lane; j < WW; j += 32) {
    __builtin_amdgcn_global_load_async_to_lds_b32(
        (as1_int*)(cids + rowoff + j), (as3_int*)&s_cid[j], 0, 0);
  }
#endif
  for (int j = lane; j < WW; j += 32) {        // 480/32 = 15 uniform iters
#if !HAVE_ASYNC_LDS
    s_cid[j] = cids[rowoff + j];
#endif
    const float o = off_logits[rowoff + j];
    const float sig = 1.0f / (1.0f + __expf(-o));
    s_x[j]    = (_Float16)((float)j + sig);
    s_z[j]    = (_Float16)height[rowoff + j];
    s_one[j]  = (_Float16)1.0f;
    s_zero[j] = (_Float16)0.0f;
  }
#if HAVE_ASYNC_LDS
#if __has_builtin(__builtin_amdgcn_s_wait_asynccnt)
  __builtin_amdgcn_s_wait_asynccnt(0);
#else
  asm volatile("s_wait_asynccnt 0" ::: "memory");
#endif
#endif
  __syncthreads();

  const int hi  = lane >> 4;   // lane half selects K sub-block
  const int m   = lane & 15;   // A row within tile / B column index
  const int m16 = m + 16;

  // Per-lane invariant base pointers (no branches in the hot loop).
  const _Float16* bbase =
      (m == 0) ? s_one : (m == 1) ? s_x : (m == 2) ? s_z : s_zero;
  bbase += hi * 16;                      // lanes>=16 hold K=16..31 of B
  const int* arun1 = s_cid + hi * 16;    // A: K run {0..7}or{16..23}
  const int* arun2 = s_cid + 8 + hi * 16;// A: K run {8..15}or{24..31}

  v8f acc0 = {};              // clusters 0..15
  v8f acc1 = {};              // clusters 16..31

  for (int p0 = 0; p0 < WW; p0 += 32) {
    const v16h  bb = *(const v16h*)(bbase + p0);   // ds_load_b128
    const int8v r1 = *(const int8v*)(arun1 + p0);  // 2x ds_load_b128
    const int8v r2 = *(const int8v*)(arun2 + p0);
    v16h a0, a1;
#pragma unroll
    for (int j = 0; j < 8; ++j) {
      a0[j]     = (_Float16)((r1[j] == m)   ? 1.0f : 0.0f);
      a0[8 + j] = (_Float16)((r2[j] == m)   ? 1.0f : 0.0f);
      a1[j]     = (_Float16)((r1[j] == m16) ? 1.0f : 0.0f);
      a1[8 + j] = (_Float16)((r2[j] == m16) ? 1.0f : 0.0f);
    }
    acc0 = __builtin_amdgcn_wmma_f32_16x16x32_f16(false, a0, false, bb,
                                                  (short)0, acc0, false, false);
    acc1 = __builtin_amdgcn_wmma_f32_16x16x32_f16(false, a1, false, bb,
                                                  (short)0, acc1, false, false);
  }

  // D 16x16 f32 layout: VGPR i -> (M=i, N=lane) lanes 0..15, (M=i+8) hi half.
  if (m < 3) {
    float* dst = (m == 0) ? cnt_kh : (m == 1) ? sx_kh : sz_kh;
    const int kb = hi * 8;
#pragma unroll
    for (int i = 0; i < 8; ++i) {
      dst[((size_t)b * KK + (kb + i)) * HH + h]      = acc0[i];
      dst[((size_t)b * KK + (16 + kb + i)) * HH + h] = acc1[i];
    }
  }
}

// ---------------------------------------------------------------------------
// Kernel 3b: per-(b,k) validity + BEV transform + H-flip; writes pts | mask.
// ---------------------------------------------------------------------------
__global__ void output_kernel(const float* __restrict__ cnt_kh,
                              const float* __restrict__ sx_kh,
                              const float* __restrict__ sz_kh,
                              const int* __restrict__ counts,
                              float* __restrict__ out) {
  const int bk = blockIdx.x;
  const int b = bk / KK;
  const int k = bk % KK;
  const int tid = threadIdx.x;       // tid = original row index i

  __shared__ int wred[10];
  __shared__ int nrows_sh;

  const size_t base = ((size_t)b * KK + k) * HH;
  const float cnt = cnt_kh[base + tid];
  const int has = (cnt > 0.0f) ? 1 : 0;

  int s = has;
#pragma unroll
  for (int off = 16; off >= 1; off >>= 1) s += __shfl_xor(s, off, 32);
  const int lane = tid & 31, wave = tid >> 5;
  if (lane == 0) wred[wave] = s;
  __syncthreads();
  if (tid == 0) {
    int tot = 0;
    for (int w = 0; w < 10; ++w) tot += wred[w];
    nrows_sh = tot;
  }
  __syncthreads();

  const bool lane_ok = (counts[b * KK + k] >= MIN_CLUSTER) && (nrows_sh >= 2);
  const bool mk = (has != 0) && lane_ok;

  const float denom = fmaxf(cnt, 1.0f);
  const float mean_x = sx_kh[base + tid] / denom;
  const float mean_z = sz_kh[base + tid] / denom;
  const float x = MAX_X - (float)tid * MPP0;
  const float y = -(mean_x * MPP1 - 240.0f * MPP1);   // W//2 = 240

  const int hout = HH - 1 - tid;                      // flip along H
  const size_t pidx = (base + hout) * 3;
  out[pidx + 0] = mk ? x : 0.0f;
  out[pidx + 1] = mk ? y : 0.0f;
  out[pidx + 2] = mk ? mean_z : 0.0f;
  out[(size_t)BB * KK * HH * 3 + base + hout] = mk ? 1.0f : 0.0f;
}

// ---------------------------------------------------------------------------
// Launcher
// ---------------------------------------------------------------------------
extern "C" void kernel_launch(void* const* d_in, const int* in_sizes, int n_in,
                              void* d_out, int out_size, void* d_ws, size_t ws_size,
                              hipStream_t stream) {
  const float* seg  = (const float*)d_in[0];
  const float* emb  = (const float*)d_in[1];
  const float* offl = (const float*)d_in[2];
  const float* hgt  = (const float*)d_in[3];
  (void)in_sizes; (void)n_in; (void)out_size; (void)ws_size;
  // d_in[4] intrinsic / d_in[5] extrinsic are unused by the reference output.

  char* ws = (char*)d_ws;
  size_t off = 0;
  auto walloc = [&](size_t bytes) -> void* {
    void* p = ws + off;
    off += (bytes + 255) & ~(size_t)255;
    return p;
  };
  int*    cids   = (int*)   walloc((size_t)BB * PP * sizeof(int));
  int*    cidx   = (int*)   walloc((size_t)BB * PP * sizeof(int));
  float4* cemb   = (float4*)walloc((size_t)BB * PP * sizeof(float4));
  int*    nvalid = (int*)   walloc((size_t)BB * sizeof(int));
  int*    counts = (int*)   walloc((size_t)BB * KK * sizeof(int));
  float*  cnt_kh = (float*) walloc((size_t)BB * KK * HH * sizeof(float));
  float*  sx_kh  = (float*) walloc((size_t)BB * KK * HH * sizeof(float));
  float*  sz_kh  = (float*) walloc((size_t)BB * KK * HH * sizeof(float));

  compact_kernel<<<BB, 256, 0, stream>>>(seg, emb, cids, cidx, cemb, nvalid);
  cluster_kernel<<<BB, 32, 0, stream>>>(nvalid, cidx, cemb, cids, counts);
  segsum_wmma_kernel<<<BB * HH, 32, 0, stream>>>(cids, offl, hgt,
                                                 cnt_kh, sx_kh, sz_kh);
  output_kernel<<<BB * KK, HH, 0, stream>>>(cnt_kh, sx_kh, sz_kh, counts,
                                            (float*)d_out);
}
